// SGCNN_85323820302957
// MI455X (gfx1250) — matrix operand order repaired and checked by reading, not compiled, not run on hardware
//
#include <hip/hip_runtime.h>

typedef unsigned short u16;
typedef unsigned int   u32;
typedef __bf16 bfx16 __attribute__((ext_vector_type(16)));
typedef float  fx8   __attribute__((ext_vector_type(8)));

#define N_NODES 40000
#define N_EDGES 640000
#define N_GRAPHS 256

union Frag { bfx16 v; uint4 q[2]; };

__device__ inline u16 f2bf(float f) {
    u32 u = __float_as_uint(f);
    u += 0x7fffu + ((u >> 16) & 1u);          // round-to-nearest-even
    return (u16)(u >> 16);
}
__device__ inline float bf2f(u16 h) { return __uint_as_float(((u32)h) << 16); }
__device__ inline float sigm(float x) { return 1.f / (1.f + __expf(-x)); }
__device__ inline float soft(float x) { return fmaxf(x, 0.f) + log1pf(__expf(-fabsf(x))); }
__device__ inline float silu(float x) { return x * sigm(x); }

// ---------------- prep kernels ----------------

// f32 [R,C] -> bf16 [R,Cp] zero-padded rows
__global__ void pack_rows(const float* __restrict__ X, int R, int C, int Cp,
                          u16* __restrict__ out) {
    int idx = blockIdx.x * 256 + threadIdx.x;
    if (idx >= R * Cp) return;
    int r = idx / Cp, c = idx - r * Cp;
    out[idx] = (c < C) ? f2bf(X[(size_t)r * C + c]) : (u16)0;
}

// Pack W [Kreal,64] f32 into WMMA B-fragment order (bf16), K padded to KC*32.
// Dense 16-bit B 32x16 layout: lane group g holds K = kb + 8*g + (j&7) + 16*(j>>3).
__global__ void pack_wfrag(const float* __restrict__ W, int Kreal, int KC,
                           u16* __restrict__ out) {
    int idx = blockIdx.x * 256 + threadIdx.x;
    int total = KC * 4 * 32 * 16;
    if (idx >= total) return;
    int j    = idx & 15;
    int lane = (idx >> 4) & 31;
    int f    = idx >> 9;            // f = kc*4 + nt
    int kc   = f >> 2, nt = f & 3;
    int K = kc * 32 + 8 * (lane >> 4) + (j & 7) + ((j >> 3) << 4);
    int n = nt * 16 + (lane & 15);
    float v = (K < Kreal) ? W[(size_t)K * 64 + n] : 0.f;
    out[idx] = f2bf(v);
}

// ---------------- WMMA GEMMs ----------------

// Edge conv: per wave 2 x 16-edge M-tiles (M=32) so each B fragment feeds two
// WMMAs. A = gathered [h[src]|h[dst]|efeat] (K=192), two weight matrices ->
// z_mlp, z_scr bf16 + fused BN column stats. Block = 8 waves = 256 edges.
__global__ __launch_bounds__(256) void gemm_edge(
    const u16* __restrict__ hbf, const u16* __restrict__ e64,
    const int* __restrict__ src, const int* __restrict__ dst,
    const u16* __restrict__ wfA, const u16* __restrict__ wfB,
    u16* __restrict__ zm, u16* __restrict__ zs,
    float* __restrict__ statsA, float* __restrict__ statsB) {
    __shared__ float sred[4][64];
    int tid = threadIdx.x;
    ((float*)sred)[tid] = 0.f;
    __syncthreads();
    int wave = tid >> 5, lane = tid & 31, g = lane >> 4, lr = lane & 15;
    int ebase = blockIdx.x * 256 + wave * 32;
    int e0 = ebase + lr, e1 = ebase + 16 + lr;
    const uint4* seg0[3] = {
        (const uint4*)(hbf + ((size_t)src[e0] << 6)),
        (const uint4*)(hbf + ((size_t)dst[e0] << 6)),
        (const uint4*)(e64 + ((size_t)e0 << 6)) };
    const uint4* seg1[3] = {
        (const uint4*)(hbf + ((size_t)src[e1] << 6)),
        (const uint4*)(hbf + ((size_t)dst[e1] << 6)),
        (const uint4*)(e64 + ((size_t)e1 << 6)) };

    fx8 zero = {0.f,0.f,0.f,0.f,0.f,0.f,0.f,0.f};
    fx8 accM[2][4] = {{zero,zero,zero,zero},{zero,zero,zero,zero}};
    fx8 accS[2][4] = {{zero,zero,zero,zero},{zero,zero,zero,zero}};

#pragma unroll
    for (int kc = 0; kc < 6; ++kc) {
        int s = kc >> 1;
        int qi = ((kc & 1) << 2) + g;       // 16B chunks per A-frag layout
        Frag a0; a0.q[0] = seg0[s][qi]; a0.q[1] = seg0[s][qi + 2];
        Frag a1; a1.q[0] = seg1[s][qi]; a1.q[1] = seg1[s][qi + 2];
#pragma unroll
        for (int nt = 0; nt < 4; ++nt) {
            int f = kc * 4 + nt;
            Frag bm; const uint4* wp = (const uint4*)wfA + (size_t)(f * 32 + lane) * 2;
            bm.q[0] = wp[0]; bm.q[1] = wp[1];
            accM[0][nt] = __builtin_amdgcn_wmma_f32_16x16x32_bf16(
                false, a0.v, false, bm.v, (short)0, accM[0][nt], false, false);
            accM[1][nt] = __builtin_amdgcn_wmma_f32_16x16x32_bf16(
                false, a1.v, false, bm.v, (short)0, accM[1][nt], false, false);
            Frag bs; const uint4* wq = (const uint4*)wfB + (size_t)(f * 32 + lane) * 2;
            bs.q[0] = wq[0]; bs.q[1] = wq[1];
            accS[0][nt] = __builtin_amdgcn_wmma_f32_16x16x32_bf16(
                false, a0.v, false, bs.v, (short)0, accS[0][nt], false, false);
            accS[1][nt] = __builtin_amdgcn_wmma_f32_16x16x32_bf16(
                false, a1.v, false, bs.v, (short)0, accS[1][nt], false, false);
        }
    }
#pragma unroll
    for (int t = 0; t < 2; ++t) {
        int tbase = ebase + 16 * t;
#pragma unroll
        for (int nt = 0; nt < 4; ++nt) {
            int col = nt * 16 + lr;
            float sm = 0.f, qm = 0.f, ss = 0.f, qs = 0.f;
#pragma unroll
            for (int r = 0; r < 8; ++r) {
                float vm = accM[t][nt][r], vs = accS[t][nt][r];
                int m = r + 8 * g;                     // C/D row layout
                size_t o = ((size_t)(tbase + m) << 6) + col;
                zm[o] = f2bf(vm); zs[o] = f2bf(vs);
                sm += vm; qm += vm * vm; ss += vs; qs += vs * vs;
            }
            atomicAdd(&sred[0][col], sm);
            atomicAdd(&sred[1][col], qm);
            atomicAdd(&sred[2][col], ss);
            atomicAdd(&sred[3][col], qs);
        }
    }
    __syncthreads();
    if (tid < 64) {
        atomicAdd(&statsA[tid],      sred[0][tid]);
        atomicAdd(&statsA[64 + tid], sred[1][tid]);
        atomicAdd(&statsB[tid],      sred[2][tid]);
        atomicAdd(&statsB[64 + tid], sred[3][tid]);
    }
}

// Node embed: A = padded node feats bf16 [R,96], one weight matrix.
__global__ __launch_bounds__(256) void gemm_node(
    const u16* __restrict__ nf, const u16* __restrict__ wf,
    u16* __restrict__ z, float* __restrict__ stats, int R) {
    __shared__ float sred[2][64];
    int tid = threadIdx.x;
    if (tid < 128) ((float*)sred)[tid] = 0.f;
    __syncthreads();
    int wave = tid >> 5, lane = tid & 31, g = lane >> 4, lr = lane & 15;
    int rbase = blockIdx.x * 128 + wave * 16;
    int row = rbase + lr;
    int rc = row < R ? row : R - 1;
    const uint4* pr = (const uint4*)(nf + (size_t)rc * 96);

    fx8 zero = {0.f,0.f,0.f,0.f,0.f,0.f,0.f,0.f};
    fx8 acc[4] = {zero,zero,zero,zero};
#pragma unroll
    for (int kc = 0; kc < 3; ++kc) {
        int qi = kc * 4 + g;
        Frag a; a.q[0] = pr[qi]; a.q[1] = pr[qi + 2];
#pragma unroll
        for (int nt = 0; nt < 4; ++nt) {
            int f = kc * 4 + nt;
            Frag b; const uint4* wp = (const uint4*)wf + (size_t)(f * 32 + lane) * 2;
            b.q[0] = wp[0]; b.q[1] = wp[1];
            acc[nt] = __builtin_amdgcn_wmma_f32_16x16x32_bf16(
                false, a.v, false, b.v, (short)0, acc[nt], false, false);
        }
    }
#pragma unroll
    for (int nt = 0; nt < 4; ++nt) {
        int col = nt * 16 + lr;
        float s = 0.f, q = 0.f;
#pragma unroll
        for (int r = 0; r < 8; ++r) {
            int ridx = rbase + r + 8 * g;
            if (ridx < R) {
                float v = acc[nt][r];
                z[((size_t)ridx << 6) + col] = f2bf(v);
                s += v; q += v * v;
            }
        }
        atomicAdd(&sred[0][col], s);
        atomicAdd(&sred[1][col], q);
    }
    __syncthreads();
    if (tid < 64) {
        atomicAdd(&stats[tid],      sred[0][tid]);
        atomicAdd(&stats[64 + tid], sred[1][tid]);
    }
}

// ---------------- BN / activation / scatter ----------------

// coef[c]=gamma*rsqrt(var+eps), coef[C+c]=beta-mean*coef[c]
__global__ void finalize_bn(const float* __restrict__ stats,
                            const float* __restrict__ gamma,
                            const float* __restrict__ beta,
                            float invN, int C, float* __restrict__ coef) {
    int c = threadIdx.x;
    if (c >= C) return;
    float mean = stats[c] * invN;
    float var  = stats[C + c] * invN - mean * mean;
    float a = gamma[c] * rsqrtf(var + 1e-5f);
    coef[c] = a;
    coef[C + c] = beta[c] - mean * a;
}

__global__ void embed_apply(const u16* __restrict__ z, const float* __restrict__ cf,
                            float* __restrict__ h, u16* __restrict__ hbf, int total) {
    int idx = blockIdx.x * 256 + threadIdx.x;
    if (idx >= total) return;
    int c = idx & 63;
    float y = silu(bf2f(z[idx]) * cf[c] + cf[64 + c]);
    h[idx] = y; hbf[idx] = f2bf(y);
}

__global__ void edge_update(const u16* __restrict__ zm, const u16* __restrict__ zs,
                            const int* __restrict__ dst, const float* __restrict__ cA,
                            const float* __restrict__ cB, float* __restrict__ agg) {
    int idx = blockIdx.x * 256 + threadIdx.x;       // E*32 threads, 2 cols each
    int e = idx >> 5, p = (idx & 31) << 1;
    u32 um = *(const u32*)(zm + ((size_t)e << 6) + p);
    u32 us = *(const u32*)(zs + ((size_t)e << 6) + p);
    float* ap = agg + ((size_t)dst[e] << 6) + p;
#pragma unroll
    for (int i = 0; i < 2; ++i) {
        int c = p + i;
        float m = bf2f((u16)(i ? (um >> 16) : (um & 0xffffu)));
        float s = bf2f((u16)(i ? (us >> 16) : (us & 0xffffu)));
        float x = m * cA[c] + cA[64 + c];
        float y = s * cB[c] + cB[64 + c];
        atomicAdd(ap + i, sigm(x) * soft(y));
    }
}

__global__ void col_stats(const float* __restrict__ X, int R, int C,
                          float* __restrict__ stats) {
    int t = threadIdx.x;
    int rp = 256 / C;
    int col = t % C, rr = t / C;
    float s = 0.f, q = 0.f;
    for (int r = blockIdx.x * rp + rr; r < R; r += gridDim.x * rp) {
        float v = X[(size_t)r * C + col];
        s += v; q += v * v;
    }
    __shared__ float ls[256], lq[256];
    ls[t] = s; lq[t] = q;
    __syncthreads();
    if (t < C) {
        for (int i = 1; i < rp; ++i) { s += ls[t + i * C]; q += lq[t + i * C]; }
        atomicAdd(&stats[col], s);
        atomicAdd(&stats[C + col], q);
    }
}

__global__ void node_update(const float* __restrict__ agg, const float* __restrict__ cf,
                            float* __restrict__ h, u16* __restrict__ hbf, int total) {
    int idx = blockIdx.x * 256 + threadIdx.x;
    if (idx >= total) return;
    int c = idx & 63;
    float y = soft(agg[idx] * cf[c] + cf[64 + c] + h[idx]);
    h[idx] = y; hbf[idx] = f2bf(y);
}

// ---------------- pooling & head ----------------

__global__ void count_k(const int* __restrict__ n2g, float* __restrict__ cnt, int N) {
    int i = blockIdx.x * 256 + threadIdx.x;
    if (i < N) atomicAdd(&cnt[n2g[i]], 1.f);
}
__global__ void pool_accum(const float* __restrict__ h, const int* __restrict__ n2g,
                           float* __restrict__ vt, int off, int total) {
    int idx = blockIdx.x * 256 + threadIdx.x;
    if (idx >= total) return;
    int n = idx >> 6, c = idx & 63;
    atomicAdd(&vt[((size_t)n2g[n] << 7) + off + c], h[idx]);
}
__global__ void pool_div(float* __restrict__ vt, const float* __restrict__ cS,
                         const float* __restrict__ cB) {
    int idx = blockIdx.x * 256 + threadIdx.x;    // 256*128
    int g = idx >> 7, c = idx & 127;
    float ct = (c < 64) ? cS[g] : cB[g];
    vt[idx] /= fmaxf(ct, 1.f);
}
__global__ void fc_forward(const float* __restrict__ X, const float* __restrict__ W,
                           float* __restrict__ Z, int K, int Nout) {
    int i = blockIdx.x, j = threadIdx.x;
    if (j >= Nout) return;
    float acc = 0.f;
    for (int k = 0; k < K; ++k) acc += X[(size_t)i * K + k] * W[(size_t)k * Nout + j];
    Z[(size_t)i * Nout + j] = acc;
}
__global__ void fc_apply(const float* __restrict__ Z, const float* __restrict__ cf,
                         float* __restrict__ out, int C, int total) {
    int idx = blockIdx.x * 256 + threadIdx.x;
    if (idx >= total) return;
    int c = idx % C;
    out[idx] = silu(Z[idx] * cf[c] + cf[C + c]);
}
__global__ void pred_k(const float* __restrict__ X, const float* __restrict__ W,
                       const float* __restrict__ b, float* __restrict__ out) {
    int i = threadIdx.x;
    if (i >= 256) return;
    float acc = b[0];
    for (int k = 0; k < 64; ++k) acc += X[(size_t)i * 64 + k] * W[k];
    out[i] = acc;
}

// ---------------- host orchestration ----------------

static inline size_t alignup(size_t x) { return (x + 255) & ~(size_t)255; }

extern "C" void kernel_launch(void* const* d_in, const int* in_sizes, int n_in,
                              void* d_out, int out_size, void* d_ws, size_t ws_size,
                              hipStream_t stream) {
    (void)in_sizes; (void)n_in; (void)out_size; (void)ws_size;
    char* ws = (char*)d_ws;
    size_t off = 0;
    auto alloc = [&](size_t bytes) -> void* { void* p = ws + off; off = alignup(off + bytes); return p; };

    float* h    = (float*)alloc((size_t)N_NODES * 64 * 4);
    u16*   hbf  = (u16*)  alloc((size_t)N_NODES * 64 * 2);
    u16*   nf96 = (u16*)  alloc((size_t)N_NODES * 96 * 2);
    u16*   e64b = (u16*)  alloc((size_t)N_EDGES * 64 * 2);
    u16*   zm   = (u16*)  alloc((size_t)N_EDGES * 64 * 2);
    u16*   zs   = (u16*)  alloc((size_t)N_EDGES * 64 * 2);
    float* agg  = (float*)alloc((size_t)N_NODES * 64 * 4);
    float* vt   = (float*)alloc(256 * 128 * 4);
    float* fcz  = (float*)alloc(256 * 128 * 4);
    float* fca  = (float*)alloc(256 * 128 * 4);
    float* cntS = (float*)alloc(256 * 4);
    float* cntB = (float*)alloc(256 * 4);
    float* stA  = (float*)alloc(128 * 4);
    float* stB  = (float*)alloc(128 * 4);
    float* cfA  = (float*)alloc(128 * 4);
    float* cfB  = (float*)alloc(128 * 4);
    float* nst  = (float*)alloc(128 * 4);
    float* ncf  = (float*)alloc(128 * 4);
    float* fst  = (float*)alloc(256 * 4);
    float* fcf  = (float*)alloc(256 * 4);
    u16*   wfA  = (u16*)  alloc(6 * 4 * 32 * 16 * 2);
    u16*   wfB  = (u16*)  alloc(6 * 4 * 32 * 16 * 2);
    u16*   wfE  = (u16*)  alloc(3 * 4 * 32 * 16 * 2);

    // params flattened in JAX pytree (sorted-key) order after the 10 data inputs:
    // 10..39 convs_b[0..2]{bn_b,bn_g,mlp.W,mlp.b,mlp.beta,mlp.g,scr.W,scr.b,scr.beta,scr.g}
    // 40..69 convs_s; 70..73 embed_b{W,b,beta,g}; 74..77 embed_s;
    // 78..85 fc0{W,b,beta,g},fc1{...}; 86 pred.W; 87 pred.b
    auto P = [&](int i) -> const float* { return (const float*)d_in[i]; };

    hipMemsetAsync(vt, 0, 256 * 128 * 4, stream);
    hipMemsetAsync(cntS, 0, 256 * 4, stream);
    hipMemsetAsync(cntB, 0, 256 * 4, stream);

    for (int br = 0; br < 2; ++br) {
        const float* vf  = P(br ? 5 : 0);
        const float* ef  = P(br ? 6 : 1);
        const int* srcI  = (const int*)d_in[br ? 7 : 2];
        const int* dstI  = (const int*)d_in[br ? 8 : 3];
        const int* n2g   = (const int*)d_in[br ? 9 : 4];
        int embedIdx = br ? 70 : 74;
        int convBase = br ? 10 : 40;

        pack_rows<<<(N_NODES * 96) / 256, 256, 0, stream>>>(vf, N_NODES, 92, 96, nf96);
        pack_rows<<<(N_EDGES * 64) / 256, 256, 0, stream>>>(ef, N_EDGES, 41, 64, e64b);
        pack_wfrag<<<(3 * 4 * 32 * 16) / 256, 256, 0, stream>>>(P(embedIdx), 92, 3, wfE);
        hipMemsetAsync(stA, 0, 128 * 4, stream);
        gemm_node<<<(N_NODES + 127) / 128, 256, 0, stream>>>(nf96, wfE, zm, stA, N_NODES);
        finalize_bn<<<1, 64, 0, stream>>>(stA, P(embedIdx + 3), P(embedIdx + 2),
                                          1.f / N_NODES, 64, cfA);
        embed_apply<<<(N_NODES * 64) / 256, 256, 0, stream>>>(zm, cfA, h, hbf, N_NODES * 64);

        for (int c = 0; c < 3; ++c) {
            int cv = convBase + 10 * c;
            pack_wfrag<<<(6 * 4 * 32 * 16) / 256, 256, 0, stream>>>(P(cv + 2), 169, 6, wfA);
            pack_wfrag<<<(6 * 4 * 32 * 16) / 256, 256, 0, stream>>>(P(cv + 6), 169, 6, wfB);
            hipMemsetAsync(stA, 0, 128 * 4, stream);
            hipMemsetAsync(stB, 0, 128 * 4, stream);
            hipMemsetAsync(agg, 0, (size_t)N_NODES * 64 * 4, stream);
            gemm_edge<<<N_EDGES / 256, 256, 0, stream>>>(hbf, e64b, srcI, dstI,
                                                         wfA, wfB, zm, zs, stA, stB);
            finalize_bn<<<1, 64, 0, stream>>>(stA, P(cv + 5), P(cv + 4), 1.f / N_EDGES, 64, cfA);
            finalize_bn<<<1, 64, 0, stream>>>(stB, P(cv + 9), P(cv + 8), 1.f / N_EDGES, 64, cfB);
            edge_update<<<(N_EDGES * 32) / 256, 256, 0, stream>>>(zm, zs, dstI, cfA, cfB, agg);
            hipMemsetAsync(nst, 0, 128 * 4, stream);
            col_stats<<<256, 256, 0, stream>>>(agg, N_NODES, 64, nst);
            finalize_bn<<<1, 64, 0, stream>>>(nst, P(cv + 1), P(cv + 0), 1.f / N_NODES, 64, ncf);
            node_update<<<(N_NODES * 64) / 256, 256, 0, stream>>>(agg, ncf, h, hbf, N_NODES * 64);
        }
        count_k<<<(N_NODES + 255) / 256, 256, 0, stream>>>(n2g, br ? cntB : cntS, N_NODES);
        pool_accum<<<(N_NODES * 64) / 256, 256, 0, stream>>>(h, n2g, vt, br * 64, N_NODES * 64);
    }
    pool_div<<<128, 256, 0, stream>>>(vt, cntS, cntB);

    fc_forward<<<256, 128, 0, stream>>>(vt, P(78), fcz, 128, 128);
    hipMemsetAsync(fst, 0, 256 * 4, stream);
    col_stats<<<8, 256, 0, stream>>>(fcz, 256, 128, fst);
    finalize_bn<<<1, 128, 0, stream>>>(fst, P(81), P(80), 1.f / 256, 128, fcf);
    fc_apply<<<(256 * 128) / 256, 256, 0, stream>>>(fcz, fcf, fca, 128, 256 * 128);

    fc_forward<<<256, 64, 0, stream>>>(fca, P(82), fcz, 128, 64);
    hipMemsetAsync(fst, 0, 256 * 4, stream);
    col_stats<<<8, 256, 0, stream>>>(fcz, 256, 64, fst);
    finalize_bn<<<1, 64, 0, stream>>>(fst, P(85), P(84), 1.f / 256, 64, fcf);
    fc_apply<<<(256 * 64) / 256, 256, 0, stream>>>(fcz, fcf, fca, 64, 256 * 64);

    pred_k<<<1, 256, 0, stream>>>(fca, P(86), P(87), (float*)d_out);
}
